// SelfAttention_10539849744802
// MI455X (gfx1250) — matrix operand order repaired and checked by reading, compiled, not run on hardware
//
#include <hip/hip_runtime.h>
#include <hip/hip_bf16.h>

#define D_MODEL 1024
#define SEQ     2048
#define NH      16
#define HD      64
#define BATCH   2
#define ROWS    (BATCH * SEQ)   // 4096

typedef __bf16 bf16_t;
typedef __bf16 v16bf __attribute__((ext_vector_type(16)));
typedef float  v8f   __attribute__((ext_vector_type(8)));

// ---------------------------------------------------------------------------
// WMMA fragment loaders (wave32 layouts per CDNA5 ISA 7.12.2)
//
// 16-bit A (16x32): lane L holds row M=L%16; with half=L/16, its 16 elements
// are K = half*8 + [0..7] then K = 16 + half*8 + [0..7]  (two 16B spans).
//
// 16-bit B (32x16): lane L holds col N=L%16; its 16 elements are
// K = (L/16)*8 + [0..7] then K = 16 + (L/16)*8 + [0..7] (two 16B spans). For
// B = Xᵀ with X row-major [N][K], those spans are contiguous in X's row N.
// ---------------------------------------------------------------------------

__device__ __forceinline__ v16bf load_a_bf16(const bf16_t* __restrict__ base,
                                             int ld, int m, int half, int k0) {
  union { v16bf v; uint4 u[2]; } x;
  const bf16_t* row = base + (size_t)m * ld + k0;
  x.u[0] = *(const uint4*)(row + half * 8);
  x.u[1] = *(const uint4*)(row + 16 + half * 8);
  return x.v;
}

__device__ __forceinline__ v16bf load_bT_bf16(const bf16_t* __restrict__ base,
                                              int ld, int n, int grp, int k0) {
  union { v16bf v; uint4 u[2]; } x;
  const bf16_t* row = base + (size_t)n * ld + k0;
  x.u[0] = *(const uint4*)(row + grp * 8);
  x.u[1] = *(const uint4*)(row + 16 + grp * 8);
  return x.v;
}

#define WMMA_BF16(A, B, C) \
  __builtin_amdgcn_wmma_f32_16x16x32_bf16(false, (A), false, (B), (short)0, (C), false, false)

// ---------------------------------------------------------------------------
// f32 -> bf16 bulk convert, 4 elements/thread (n must be a multiple of 4)
// ---------------------------------------------------------------------------
__global__ void __launch_bounds__(256)
f32_to_bf16(const float* __restrict__ src, bf16_t* __restrict__ dst, int n) {
  int i = (blockIdx.x * 256 + threadIdx.x) * 4;
  if (i >= n) return;
  float4 f = *(const float4*)(src + i);
  union { bf16_t h[4]; uint2 u; } p;
  p.h[0] = (bf16_t)f.x; p.h[1] = (bf16_t)f.y;
  p.h[2] = (bf16_t)f.z; p.h[3] = (bf16_t)f.w;
  *(uint2*)(dst + i) = p.u;
}

// ---------------------------------------------------------------------------
// Vt[b,h,d,s] = bf16(H[b,s,h*64+d])   (per-head transposed V)
// ---------------------------------------------------------------------------
__global__ void __launch_bounds__(256)
build_vt(const float* __restrict__ H, bf16_t* __restrict__ Vt, int total) {
  int i = blockIdx.x * blockDim.x + threadIdx.x;
  if (i >= total) return;
  int s = i % SEQ;
  int d = (i / SEQ) % HD;
  int h = (i / (SEQ * HD)) % NH;
  int b = i / (SEQ * HD * NH);
  Vt[i] = (bf16_t)H[((size_t)b * SEQ + s) * D_MODEL + h * HD + d];
}

// ---------------------------------------------------------------------------
// Y(bf16)[ROWS,D] = (X(bf16) @ W(bf16)^T + bias) * scale
// One wave computes a 32x64 tile: 8 accumulators, 32 K-steps, 8 WMMAs/step.
// ---------------------------------------------------------------------------
__global__ void __launch_bounds__(256)
proj_kernel(const bf16_t* __restrict__ X, const bf16_t* __restrict__ W,
            const float* __restrict__ bias, bf16_t* __restrict__ Y, float scale) {
  const int lane = threadIdx.x & 31;
  const int warp = threadIdx.x >> 5;
  const int w    = blockIdx.x * 8 + warp;          // 2048 waves total
  const int nt_n = D_MODEL / 64;                   // 16
  const int m0   = (w / nt_n) * 32;
  const int n0   = (w % nt_n) * 64;
  const int lm   = lane & 15;
  const int half = lane >> 4;

  v8f acc[2][4] = {};
  for (int k0 = 0; k0 < D_MODEL; k0 += 32) {
    v16bf a0 = load_a_bf16(X, D_MODEL, m0 + lm, half, k0);
    v16bf a1 = load_a_bf16(X, D_MODEL, m0 + 16 + lm, half, k0);
#pragma unroll
    for (int f = 0; f < 4; ++f) {
      v16bf b = load_bT_bf16(W, D_MODEL, n0 + f * 16 + lm, half, k0);
      acc[0][f] = WMMA_BF16(a0, b, acc[0][f]);
      acc[1][f] = WMMA_BF16(a1, b, acc[1][f]);
    }
  }
#pragma unroll
  for (int mi = 0; mi < 2; ++mi)
#pragma unroll
    for (int f = 0; f < 4; ++f) {
      float bv = bias[n0 + f * 16 + lm];
#pragma unroll
      for (int r = 0; r < 8; ++r) {
        int row = m0 + mi * 16 + r + 8 * half;
        Y[(size_t)row * D_MODEL + n0 + f * 16 + lm] =
            (bf16_t)((acc[mi][f][r] + bv) * scale);
      }
    }
}

// ---------------------------------------------------------------------------
// out(f32)[ROWS,D] = A(bf16) @ Wo(bf16)^T + bo   (32x64 tile per wave)
// ---------------------------------------------------------------------------
__global__ void __launch_bounds__(256)
oproj_kernel(const bf16_t* __restrict__ X, const bf16_t* __restrict__ W,
             const float* __restrict__ bias, float* __restrict__ Y) {
  const int lane = threadIdx.x & 31;
  const int warp = threadIdx.x >> 5;
  const int w    = blockIdx.x * 8 + warp;
  const int nt_n = D_MODEL / 64;
  const int m0   = (w / nt_n) * 32;
  const int n0   = (w % nt_n) * 64;
  const int lm   = lane & 15;
  const int half = lane >> 4;

  v8f acc[2][4] = {};
  for (int k0 = 0; k0 < D_MODEL; k0 += 32) {
    v16bf a0 = load_a_bf16(X, D_MODEL, m0 + lm, half, k0);
    v16bf a1 = load_a_bf16(X, D_MODEL, m0 + 16 + lm, half, k0);
#pragma unroll
    for (int f = 0; f < 4; ++f) {
      v16bf b = load_bT_bf16(W, D_MODEL, n0 + f * 16 + lm, half, k0);
      acc[0][f] = WMMA_BF16(a0, b, acc[0][f]);
      acc[1][f] = WMMA_BF16(a1, b, acc[1][f]);
    }
  }
#pragma unroll
  for (int mi = 0; mi < 2; ++mi)
#pragma unroll
    for (int f = 0; f < 4; ++f) {
      float bv = bias[n0 + f * 16 + lm];
#pragma unroll
      for (int r = 0; r < 8; ++r) {
        int row = m0 + mi * 16 + r + 8 * half;
        Y[(size_t)row * D_MODEL + n0 + f * 16 + lm] = acc[mi][f][r] + bv;
      }
    }
}

// ---------------------------------------------------------------------------
// Flash attention. One wave per (b, h, 16-query tile). Q is pre-scaled by
// 1/sqrt(HD) in the projection. Per 32-key block: 4 score WMMAs, online
// softmax (shfl-xor within the 16-lane half matching the C-layout row
// grouping), P -> LDS -> A-layout, 4 PV WMMAs against transposed-V fragments.
// ---------------------------------------------------------------------------
__global__ void __launch_bounds__(256)
attn_kernel(const bf16_t* __restrict__ Q, const bf16_t* __restrict__ K,
            const bf16_t* __restrict__ Vt, bf16_t* __restrict__ O) {
  __shared__ __align__(16) bf16_t lds[8][16][32];

  const int lane = threadIdx.x & 31;
  const int warp = threadIdx.x >> 5;
  const int w    = blockIdx.x * 8 + warp;          // 4096 waves = 2*16*128
  const int qtpb = SEQ / 16;                       // 128 q-tiles per (b,h)
  const int b    = w / (NH * qtpb);
  const int h    = (w / qtpb) % NH;
  const int q0   = (w % qtpb) * 16;
  const int lm   = lane & 15;
  const int half = lane >> 4;

  const bf16_t* Qh = Q  + (size_t)b * SEQ * D_MODEL + h * HD;
  const bf16_t* Kh = K  + (size_t)b * SEQ * D_MODEL + h * HD;
  const bf16_t* Vh = Vt + (size_t)(b * NH + h) * HD * SEQ;

  v16bf qa0 = load_a_bf16(Qh, D_MODEL, q0 + lm, half, 0);
  v16bf qa1 = load_a_bf16(Qh, D_MODEL, q0 + lm, half, 32);

  v8f oacc[4] = {};
  float rmax[8], rsum[8];
#pragma unroll
  for (int r = 0; r < 8; ++r) { rmax[r] = -3.0e38f; rsum[r] = 0.0f; }

  for (int n0 = 0; n0 < SEQ; n0 += 32) {
    v8f s[2] = {};
#pragma unroll
    for (int j = 0; j < 2; ++j) {
      v16bf kb0 = load_bT_bf16(Kh, D_MODEL, n0 + j * 16 + lm, half, 0);
      s[j] = WMMA_BF16(qa0, kb0, s[j]);
      v16bf kb1 = load_bT_bf16(Kh, D_MODEL, n0 + j * 16 + lm, half, 32);
      s[j] = WMMA_BF16(qa1, kb1, s[j]);
    }

#pragma unroll
    for (int r = 0; r < 8; ++r) {
      float s0 = s[0][r];
      float s1 = s[1][r];
      float mloc = fmaxf(s0, s1);
#pragma unroll
      for (int mask = 1; mask < 16; mask <<= 1)
        mloc = fmaxf(mloc, __shfl_xor(mloc, mask, 32));
      float mnew = fmaxf(rmax[r], mloc);
      float corr = __expf(rmax[r] - mnew);
      float p0 = __expf(s0 - mnew);
      float p1 = __expf(s1 - mnew);
      float sl = p0 + p1;
#pragma unroll
      for (int mask = 1; mask < 16; mask <<= 1)
        sl += __shfl_xor(sl, mask, 32);
      rsum[r] = rsum[r] * corr + sl;
      rmax[r] = mnew;
#pragma unroll
      for (int f = 0; f < 4; ++f) oacc[f][r] *= corr;
      lds[warp][r + 8 * half][lm]      = (bf16_t)p0;
      lds[warp][r + 8 * half][16 + lm] = (bf16_t)p1;
    }
    __syncthreads();

    union { v16bf v; uint4 u[2]; } pa;
    pa.u[0] = *(const uint4*)&lds[warp][lm][half * 8];
    pa.u[1] = *(const uint4*)&lds[warp][lm][16 + half * 8];
    __syncthreads();

#pragma unroll
    for (int f = 0; f < 4; ++f) {
      v16bf vb = load_bT_bf16(Vh, SEQ, f * 16 + lm, half, n0);
      oacc[f] = WMMA_BF16(pa.v, vb, oacc[f]);
    }
  }

#pragma unroll
  for (int f = 0; f < 4; ++f)
#pragma unroll
    for (int r = 0; r < 8; ++r) {
      float v = oacc[f][r] / rsum[r];
      O[((size_t)b * SEQ + q0 + r + 8 * half) * D_MODEL + h * HD + f * 16 + lm] =
          (bf16_t)v;
    }
}

// ---------------------------------------------------------------------------
extern "C" void kernel_launch(void* const* d_in, const int* in_sizes, int n_in,
                              void* d_out, int out_size, void* d_ws, size_t ws_size,
                              hipStream_t stream) {
  const float* H  = (const float*)d_in[0];
  const float* Wq = (const float*)d_in[1];
  const float* bq = (const float*)d_in[2];
  const float* Wk = (const float*)d_in[3];
  const float* bk = (const float*)d_in[4];
  const float* Wo = (const float*)d_in[5];
  const float* bo = (const float*)d_in[6];
  float* out = (float*)d_out;

  // workspace layout (bf16):
  //   Hb, Qb, Kb, Vt, AO : ROWS*D_MODEL each (8 MiB)   -> 40 MiB
  //   Wqb, Wkb, Wob      : D_MODEL*D_MODEL each (2 MiB) -> 6 MiB
  bf16_t* Hb  = (bf16_t*)d_ws;
  bf16_t* Qb  = Hb  + (size_t)ROWS * D_MODEL;
  bf16_t* Kb  = Qb  + (size_t)ROWS * D_MODEL;
  bf16_t* Vt  = Kb  + (size_t)ROWS * D_MODEL;
  bf16_t* AO  = Vt  + (size_t)ROWS * D_MODEL;
  bf16_t* Wqb = AO  + (size_t)ROWS * D_MODEL;
  bf16_t* Wkb = Wqb + (size_t)D_MODEL * D_MODEL;
  bf16_t* Wob = Wkb + (size_t)D_MODEL * D_MODEL;

  const int nH = ROWS * D_MODEL;        // 4 Mi elements
  const int nW = D_MODEL * D_MODEL;     // 1 Mi elements
  f32_to_bf16<<<nH / 4 / 256, 256, 0, stream>>>(H,  Hb,  nH);
  f32_to_bf16<<<nW / 4 / 256, 256, 0, stream>>>(Wq, Wqb, nW);
  f32_to_bf16<<<nW / 4 / 256, 256, 0, stream>>>(Wk, Wkb, nW);
  f32_to_bf16<<<nW / 4 / 256, 256, 0, stream>>>(Wo, Wob, nW);

  build_vt<<<nH / 256, 256, 0, stream>>>(H, Vt, nH);

  const float qscale = 0.125f;          // 1/sqrt(HD)
  const int gemm_blocks = (ROWS / 32) * (D_MODEL / 64) / 8;  // 256
  proj_kernel<<<gemm_blocks, 256, 0, stream>>>(Hb, Wqb, bq, Qb, qscale);
  proj_kernel<<<gemm_blocks, 256, 0, stream>>>(Hb, Wkb, bk, Kb, 1.0f);

  const int attn_blocks = (BATCH * NH * (SEQ / 16)) / 8;     // 512
  attn_kernel<<<attn_blocks, 256, 0, stream>>>(Qb, Kb, Vt, AO);

  oproj_kernel<<<gemm_blocks, 256, 0, stream>>>(AO, Wob, bo, out);
}
